// EncodeProcessDecode_23270132810477
// MI455X (gfx1250) — matrix operand order repaired and checked by reading, not compiled, not run on hardware
//
#include <hip/hip_runtime.h>
#include <hip/hip_bf16.h>

// ---------------------------------------------------------------------------
// EncodeProcessDecode GraphNet on MI455X (gfx1250), wave32 + WMMA f16.
// Latents stored f16 (halves HBM traffic; matches WMMA operand precision).
// All MLPs run on v_wmma_f32_16x16x32_f16 over 16-row tiles (1 tile / wave).
// Weights are pre-swizzled into per-lane fragment order so each B fragment is
// a single contiguous 32B load per lane (2x global_load_b128, coalesced).
// ---------------------------------------------------------------------------

typedef __attribute__((ext_vector_type(16))) _Float16 v16h;
typedef __attribute__((ext_vector_type(8)))  float    v8f;
typedef __attribute__((ext_vector_type(4)))  _Float16 half4v;

#define E_EDGES  800000
#define N_NODES  50000
#define N_GRAPHS 64
#define NSTEPS   4

static __device__ __forceinline__ v8f wmma16(v16h a, v16h b, v8f c) {
  return __builtin_amdgcn_wmma_f32_16x16x32_f16(false, a, false, b, (short)0, c,
                                                false, false);
}

// A fragment (16x32 f16, M x K) from LDS row-major [16][ldk].
// ISA layout: lane<16 row M=lane holds K {0..7,16..23}; lane>=16 holds {8..15,24..31}.
// Each 8-element group is contiguous in LDS -> lowers to 2x ds_load_b128.
static __device__ __forceinline__ v16h load_a_frag(const _Float16* src, int ldk,
                                                   int kbase, int lane) {
  const int hp = lane >> 4, m = lane & 15;
  v16h a;
#pragma unroll
  for (int i = 0; i < 16; ++i) {
    int k = kbase + ((i >> 3) << 4) + (hp << 3) + (i & 7);
    a[i] = src[m * ldk + k];
  }
  return a;
}

// B fragment from pre-swizzled weights: layout [kchunk][lane][16 halves].
// One contiguous 32B read per lane.
static __device__ __forceinline__ v16h load_b_swz(const _Float16* wswz,
                                                  int kchunk, int lane) {
  return *(const v16h*)(wswz + (((kchunk << 5) + lane) << 4));
}

// Two-layer MLP (KPAD -> 16 -> 16, ReLU after each) on one 16-row tile.
// ldsin: [16][KPAD] f16 input rows. ldsh: [16][32] f16 scratch.
// Result in C/D layout: element v of lane = out[v + 8*(lane>>4)][lane&15].
template <int KPAD>
static __device__ __forceinline__ v8f mlp_tile(const _Float16* ldsin,
                                               _Float16* ldsh,
                                               const _Float16* w1s,
                                               const float* b1,
                                               const _Float16* w2s,
                                               const float* b2, int lane) {
  v8f acc = {};
#pragma unroll
  for (int kb = 0; kb < KPAD; kb += 32)
    acc = wmma16(load_a_frag(ldsin, KPAD, kb, lane), load_b_swz(w1s, kb >> 5, lane),
                 acc);
  const int n = lane & 15, hi = lane >> 4;
  const float bb = b1[n];
#pragma unroll
  for (int v = 0; v < 8; ++v) {
    float x = acc[v] + bb;
    x = x > 0.f ? x : 0.f;
    const int m = v + (hi << 3);
    ldsh[m * 32 + n] = (_Float16)x;
    ldsh[m * 32 + 16 + n] = (_Float16)0.f;
  }
  __syncthreads();
  v8f z = {};
  v8f out = wmma16(load_a_frag(ldsh, 32, 0, lane), load_b_swz(w2s, 0, lane), z);
  const float b2n = b2[n];
#pragma unroll
  for (int v = 0; v < 8; ++v) {
    float x = out[v] + b2n;
    out[v] = x > 0.f ? x : 0.f;
  }
  return out;
}

// ---------------------------------------------------------------------------
// Utility kernels
// ---------------------------------------------------------------------------

__global__ void fill0_kernel(float* __restrict__ p, int n) {
  int i = blockIdx.x * blockDim.x + threadIdx.x;
  if (i < n) p[i] = 0.f;
}

// f32 [ksrc][16] row-major -> f16 swizzled fragments [kpad/32][32][16]:
// dst[(c*32+lane)*16+i] = w1[k][n], k = c*32 + (i>>3)*16 + (lane>>4)*8 + (i&7), n = lane&15
// (zero-padded K rows).
__global__ void convert_swz_kernel(const float* __restrict__ src,
                                   _Float16* __restrict__ dst, int ksrc,
                                   int kpad) {
  for (int idx = threadIdx.x; idx < kpad * 16; idx += blockDim.x) {
    const int c = idx >> 9;          // 32-K chunk (512 halves per chunk)
    const int rem = idx & 511;
    const int lane = rem >> 4;
    const int i = rem & 15;
    const int n = lane & 15;
    const int k = (c << 5) + ((i >> 3) << 4) + ((lane >> 4) << 3) + (i & 7);
    float v = (k < ksrc) ? src[k * 16 + n] : 0.f;
    dst[idx] = (_Float16)v;
  }
}

__global__ void count_edges_kernel(const int* __restrict__ receivers,
                                   const int* __restrict__ edge_graph, int n,
                                   float* __restrict__ cnt_rcv,
                                   float* __restrict__ cnt_eg) {
  __shared__ float lds[N_GRAPHS];
  const int t = threadIdx.x;
  if (t < N_GRAPHS) lds[t] = 0.f;
  __syncthreads();
  const int i = blockIdx.x * blockDim.x + t;
  if (i < n) {
    atomicAdd(&cnt_rcv[receivers[i]], 1.f);
    atomicAdd(&lds[edge_graph[i] & (N_GRAPHS - 1)], 1.f);
  }
  __syncthreads();
  if (t < N_GRAPHS) {
    float v = lds[t];
    if (v != 0.f) atomicAdd(&cnt_eg[t], v);
  }
}

__global__ void count_nodes_kernel(const int* __restrict__ node_graph, int n,
                                   float* __restrict__ cnt_ng) {
  __shared__ float lds[N_GRAPHS];
  const int t = threadIdx.x;
  if (t < N_GRAPHS) lds[t] = 0.f;
  __syncthreads();
  const int i = blockIdx.x * blockDim.x + t;
  if (i < n) atomicAdd(&lds[node_graph[i] & (N_GRAPHS - 1)], 1.f);
  __syncthreads();
  if (t < N_GRAPHS) {
    float v = lds[t];
    if (v != 0.f) atomicAdd(&cnt_ng[t], v);
  }
}

// ---------------------------------------------------------------------------
// Encoder: rows x 32 f32 -> 16 -> 16, writes f16 latent0 and latent.
// ---------------------------------------------------------------------------
__global__ void __launch_bounds__(128) encoder_kernel(
    const float* __restrict__ x, int rows, const _Float16* __restrict__ w1s,
    const float* __restrict__ b1, const _Float16* __restrict__ w2s,
    const float* __restrict__ b2, _Float16* __restrict__ lat0,
    _Float16* __restrict__ lat) {
  __shared__ _Float16 lds_in[4][16 * 32];
  __shared__ _Float16 lds_h[4][16 * 32];
  const int t = threadIdx.x, w = t >> 5, lane = t & 31;
  const int ntiles = rows / 16;
  int tile = blockIdx.x * 4 + w;
  const bool active = tile < ntiles;
  if (!active) tile = ntiles - 1;
  const int base = tile * 16;
#pragma unroll 4
  for (int r = 0; r < 16; ++r)
    lds_in[w][r * 32 + lane] = (_Float16)x[(size_t)(base + r) * 32 + lane];
  __syncthreads();
  v8f o = mlp_tile<32>(lds_in[w], lds_h[w], w1s, b1, w2s, b2, lane);
  const int n = lane & 15, hi = lane >> 4;
  if (active) {
#pragma unroll
    for (int v = 0; v < 8; ++v) {
      const int row = base + v + (hi << 3);
      const _Float16 hv = (_Float16)o[v];
      lat0[(size_t)row * 16 + n] = hv;
      lat[(size_t)row * 16 + n] = hv;
    }
  }
}

// ---------------------------------------------------------------------------
// Edge core: in=[e0|e|n_c[s]|n_c[r]|g_c[eg]] (128) -> 16 -> 16 = new e.
// Fused: e_to_n atomics, e_to_g LDS accumulation, dec_e (16->16->2).
// ---------------------------------------------------------------------------
__global__ void __launch_bounds__(128) edge_core_kernel(
    const _Float16* __restrict__ e0h, _Float16* __restrict__ eh,
    const _Float16* __restrict__ n0h, const _Float16* __restrict__ nh,
    const _Float16* __restrict__ g0h, const _Float16* __restrict__ gh,
    const int* __restrict__ senders, const int* __restrict__ receivers,
    const int* __restrict__ edge_graph, const _Float16* __restrict__ w1s,
    const float* __restrict__ b1, const _Float16* __restrict__ w2s,
    const float* __restrict__ b2, const _Float16* __restrict__ d1s,
    const float* __restrict__ db1, const float* __restrict__ d2,
    const float* __restrict__ db2, float* __restrict__ node_acc,
    float* __restrict__ gacc_e, float* __restrict__ out_e) {
  __shared__ _Float16 lds_in[4][16 * 128];
  __shared__ _Float16 lds_h[4][16 * 32];
  __shared__ float lds_d[4][16 * 16];
  __shared__ float lds_g[N_GRAPHS * 16];
  const int t = threadIdx.x, w = t >> 5, lane = t & 31;
  for (int i = t; i < N_GRAPHS * 16; i += 128) lds_g[i] = 0.f;
  const int tile = blockIdx.x * 4 + w;  // E divisible by 64: all tiles valid
  const int base = tile * 16;
  // Stage 16 rows x 128 f16 features: 8 segments of 16, lane loads 4 halves.
  const int seg = lane >> 2, off4 = (lane & 3) << 2;
#pragma unroll 4
  for (int r = 0; r < 16; ++r) {
    const int er = base + r;
    const int snd = senders[er], rcv = receivers[er], eg = edge_graph[er];
    (void)rcv;
    const int rowi = (seg < 2) ? er : ((seg < 4) ? snd : ((seg < 6) ? rcv : eg));
    const _Float16* p0 = (seg < 2) ? e0h : ((seg < 6) ? n0h : g0h);
    const _Float16* p1 = (seg < 2) ? eh : ((seg < 6) ? nh : gh);
    const _Float16* bp = ((seg & 1) ? p1 : p0) + (size_t)rowi * 16 + off4;
    *(half4v*)&lds_in[w][r * 128 + (lane << 2)] = *(const half4v*)bp;
  }
  __syncthreads();
  v8f eo = mlp_tile<128>(lds_in[w], lds_h[w], w1s, b1, w2s, b2, lane);
  const int n = lane & 15, hi = lane >> 4;
#pragma unroll
  for (int v = 0; v < 8; ++v) {
    const int m = v + (hi << 3);
    const int row = base + m;
    const float val = eo[v];
    eh[(size_t)row * 16 + n] = (_Float16)val;
    atomicAdd(&node_acc[(size_t)receivers[row] * 16 + n], val);
    atomicAdd(&lds_g[edge_graph[row] * 16 + n], val);
    lds_h[w][m * 32 + n] = (_Float16)val;
    lds_h[w][m * 32 + 16 + n] = (_Float16)0.f;
  }
  __syncthreads();
  // dec_e layer 1 via WMMA
  v8f z = {};
  v8f hd = wmma16(load_a_frag(lds_h[w], 32, 0, lane), load_b_swz(d1s, 0, lane), z);
  const float db1n = db1[n];
#pragma unroll
  for (int v = 0; v < 8; ++v) {
    float x = hd[v] + db1n;
    lds_d[w][(v + (hi << 3)) * 16 + n] = x > 0.f ? x : 0.f;
  }
  __syncthreads();
  {  // dec_e layer 2: 16 -> 2, lane -> (row=lane>>1, col=lane&1)
    const int orow = lane >> 1, oc = lane & 1;
    float s = db2[oc];
#pragma unroll
    for (int k = 0; k < 16; ++k) s += lds_d[w][orow * 16 + k] * d2[k * 2 + oc];
    out_e[(size_t)(base + orow) * 2 + oc] = s > 0.f ? s : 0.f;
  }
  __syncthreads();
  for (int i = t; i < N_GRAPHS * 16; i += 128) {
    const float v = lds_g[i];
    if (v != 0.f) atomicAdd(&gacc_e[i], v);
  }
}

// ---------------------------------------------------------------------------
// Node core: in=[n0|n|mean(e_to_n)|g0[ng]|g[ng]|pad] (96) -> 16 -> 16 = new n.
// Fused: n_to_g LDS accumulation, dec_n (16->16->1).
// ---------------------------------------------------------------------------
__global__ void __launch_bounds__(128) node_core_kernel(
    const _Float16* __restrict__ n0h, _Float16* __restrict__ nh,
    const _Float16* __restrict__ g0h, const _Float16* __restrict__ gh,
    const int* __restrict__ node_graph, const float* __restrict__ node_acc,
    const float* __restrict__ cnt_rcv, const _Float16* __restrict__ w1s,
    const float* __restrict__ b1, const _Float16* __restrict__ w2s,
    const float* __restrict__ b2, const _Float16* __restrict__ d1s,
    const float* __restrict__ db1, const float* __restrict__ d2,
    const float* __restrict__ db2, float* __restrict__ gacc_n,
    float* __restrict__ out_n) {
  __shared__ _Float16 lds_in[4][16 * 96];
  __shared__ _Float16 lds_h[4][16 * 32];
  __shared__ float lds_d[4][16 * 16];
  __shared__ float lds_g[N_GRAPHS * 16];
  const int t = threadIdx.x, w = t >> 5, lane = t & 31;
  for (int i = t; i < N_GRAPHS * 16; i += 128) lds_g[i] = 0.f;
  const int ntiles = N_NODES / 16;
  int tile = blockIdx.x * 4 + w;
  const bool active = tile < ntiles;
  if (!active) tile = ntiles - 1;
  const int base = tile * 16;
#pragma unroll 4
  for (int r = 0; r < 16; ++r) {
    const int nr = base + r;
    const int ngr = node_graph[nr];
    const float inv = 1.f / fmaxf(cnt_rcv[nr], 1.f);
#pragma unroll
    for (int j = 0; j < 3; ++j) {
      const int f = lane * 3 + j;       // 0..95
      const int s6 = f >> 4, off = f & 15;
      float val;
      if (s6 == 0)      val = (float)n0h[(size_t)nr * 16 + off];
      else if (s6 == 1) val = (float)nh[(size_t)nr * 16 + off];
      else if (s6 == 2) val = node_acc[(size_t)nr * 16 + off] * inv;
      else if (s6 == 3) val = (float)g0h[ngr * 16 + off];
      else if (s6 == 4) val = (float)gh[ngr * 16 + off];
      else              val = 0.f;      // pad 80..95
      lds_in[w][r * 96 + f] = (_Float16)val;
    }
  }
  __syncthreads();
  v8f no = mlp_tile<96>(lds_in[w], lds_h[w], w1s, b1, w2s, b2, lane);
  const int n = lane & 15, hi = lane >> 4;
  if (active) {
#pragma unroll
    for (int v = 0; v < 8; ++v) {
      const int m = v + (hi << 3);
      const int row = base + m;
      const float val = no[v];
      nh[(size_t)row * 16 + n] = (_Float16)val;
      atomicAdd(&lds_g[node_graph[row] * 16 + n], val);
    }
  }
#pragma unroll
  for (int v = 0; v < 8; ++v) {
    const int m = v + (hi << 3);
    lds_h[w][m * 32 + n] = (_Float16)no[v];
    lds_h[w][m * 32 + 16 + n] = (_Float16)0.f;
  }
  __syncthreads();
  v8f z = {};
  v8f hd = wmma16(load_a_frag(lds_h[w], 32, 0, lane), load_b_swz(d1s, 0, lane), z);
  const float db1n = db1[n];
#pragma unroll
  for (int v = 0; v < 8; ++v) {
    float x = hd[v] + db1n;
    lds_d[w][(v + (hi << 3)) * 16 + n] = x > 0.f ? x : 0.f;
  }
  __syncthreads();
  if (active && lane < 16) {
    float s = db2[0];
#pragma unroll
    for (int k = 0; k < 16; ++k) s += lds_d[w][lane * 16 + k] * d2[k];
    out_n[base + lane] = s > 0.f ? s : 0.f;
  }
  __syncthreads();
  for (int i = t; i < N_GRAPHS * 16; i += 128) {
    const float v = lds_g[i];
    if (v != 0.f) atomicAdd(&gacc_n[i], v);
  }
}

// ---------------------------------------------------------------------------
// Graph core + dec_g: 64 rows only, scalar math. One block of 64 threads.
// ---------------------------------------------------------------------------
__global__ void __launch_bounds__(64) graph_core_kernel(
    const _Float16* __restrict__ g0h, _Float16* __restrict__ gh,
    const float* __restrict__ gacc_e, const float* __restrict__ cnt_eg,
    const float* __restrict__ gacc_n, const float* __restrict__ cnt_ng,
    const float* __restrict__ w1, const float* __restrict__ b1,
    const float* __restrict__ w2, const float* __restrict__ b2,
    const float* __restrict__ dw1, const float* __restrict__ db1,
    const float* __restrict__ dw2, const float* __restrict__ db2,
    float* __restrict__ out_g) {
  const int gi = threadIdx.x;  // 0..63
  float in[64];
#pragma unroll
  for (int k = 0; k < 16; ++k) in[k] = (float)g0h[gi * 16 + k];
#pragma unroll
  for (int k = 0; k < 16; ++k) in[16 + k] = (float)gh[gi * 16 + k];
  const float ie = 1.f / fmaxf(cnt_eg[gi], 1.f);
  const float inn = 1.f / fmaxf(cnt_ng[gi], 1.f);
#pragma unroll
  for (int k = 0; k < 16; ++k) in[32 + k] = gacc_e[gi * 16 + k] * ie;
#pragma unroll
  for (int k = 0; k < 16; ++k) in[48 + k] = gacc_n[gi * 16 + k] * inn;
  float h1[16];
#pragma unroll
  for (int j = 0; j < 16; ++j) {
    float s = b1[j];
#pragma unroll
    for (int k = 0; k < 64; ++k) s += in[k] * w1[k * 16 + j];
    h1[j] = s > 0.f ? s : 0.f;
  }
  float g2[16];
#pragma unroll
  for (int j = 0; j < 16; ++j) {
    float s = b2[j];
#pragma unroll
    for (int k = 0; k < 16; ++k) s += h1[k] * w2[k * 16 + j];
    g2[j] = s > 0.f ? s : 0.f;
  }
#pragma unroll
  for (int j = 0; j < 16; ++j) gh[gi * 16 + j] = (_Float16)g2[j];
  float hd[16];
#pragma unroll
  for (int j = 0; j < 16; ++j) {
    float s = db1[j];
#pragma unroll
    for (int k = 0; k < 16; ++k) s += g2[k] * dw1[k * 16 + j];
    hd[j] = s > 0.f ? s : 0.f;
  }
#pragma unroll
  for (int c = 0; c < 3; ++c) {
    float s = db2[c];
#pragma unroll
    for (int k = 0; k < 16; ++k) s += hd[k] * dw2[k * 3 + c];
    out_g[gi * 3 + c] = s > 0.f ? s : 0.f;
  }
}

// ---------------------------------------------------------------------------
// Host side
// ---------------------------------------------------------------------------
extern "C" void kernel_launch(void* const* d_in, const int* in_sizes, int n_in,
                              void* d_out, int out_size, void* d_ws, size_t ws_size,
                              hipStream_t stream) {
  (void)in_sizes; (void)n_in; (void)out_size; (void)ws_size;
  const float* x_e = (const float*)d_in[0];
  const float* x_n = (const float*)d_in[1];
  const float* x_g = (const float*)d_in[2];
  const int* senders = (const int*)d_in[3];
  const int* receivers = (const int*)d_in[4];
  const int* edge_graph = (const int*)d_in[5];
  const int* node_graph = (const int*)d_in[6];
  // d_in[7] = num_steps (constant 4; cannot read device scalar under capture)
  // params flattened in setup_inputs() insertion order, each MLP (w1,b1,w2,b2):
  // enc_e, enc_n, enc_g, core_e, core_n, core_g, dec_e, dec_n, dec_g
  const float* P[36];
  for (int i = 0; i < 36; ++i) P[i] = (const float*)d_in[8 + i];
  // P idx: enc_e 0-3, enc_n 4-7, enc_g 8-11, core_e 12-15, core_n 16-19,
  //        core_g 20-23, dec_e 24-27, dec_n 28-31, dec_g 32-35

  // ---- workspace carving ----
  char* ws = (char*)d_ws;
  size_t off = 0;
  auto take = [&](size_t bytes) -> char* {
    char* p = ws + off;
    off = (off + bytes + 255) & ~(size_t)255;
    return p;
  };
  _Float16* e0h = (_Float16*)take((size_t)E_EDGES * 16 * 2);
  _Float16* ehh = (_Float16*)take((size_t)E_EDGES * 16 * 2);
  _Float16* n0h = (_Float16*)take((size_t)N_NODES * 16 * 2);
  _Float16* nhh = (_Float16*)take((size_t)N_NODES * 16 * 2);
  _Float16* g0h = (_Float16*)take((size_t)N_GRAPHS * 16 * 2);
  _Float16* ghh = (_Float16*)take((size_t)N_GRAPHS * 16 * 2);
  float* acc = (float*)take(((size_t)N_NODES * 16 + 2 * N_GRAPHS * 16) * 4);
  float* node_acc = acc;
  float* gacc_e = acc + (size_t)N_NODES * 16;
  float* gacc_n = gacc_e + N_GRAPHS * 16;
  float* cnt = (float*)take(((size_t)N_NODES + 2 * N_GRAPHS) * 4);
  float* cnt_rcv = cnt;
  float* cnt_eg = cnt + N_NODES;
  float* cnt_ng = cnt_eg + N_GRAPHS;
  _Float16* w_enc_e1 = (_Float16*)take(32 * 16 * 2);
  _Float16* w_enc_n1 = (_Float16*)take(32 * 16 * 2);
  _Float16* w_enc_g1 = (_Float16*)take(32 * 16 * 2);
  _Float16* w_core_e1 = (_Float16*)take(128 * 16 * 2);
  _Float16* w_core_n1 = (_Float16*)take(96 * 16 * 2);
  _Float16* w_enc_e2 = (_Float16*)take(32 * 16 * 2);
  _Float16* w_enc_n2 = (_Float16*)take(32 * 16 * 2);
  _Float16* w_enc_g2 = (_Float16*)take(32 * 16 * 2);
  _Float16* w_core_e2 = (_Float16*)take(32 * 16 * 2);
  _Float16* w_core_n2 = (_Float16*)take(32 * 16 * 2);
  _Float16* w_dec_e1 = (_Float16*)take(32 * 16 * 2);
  _Float16* w_dec_n1 = (_Float16*)take(32 * 16 * 2);

  // ---- one-time prep: swizzled f16 weights, segment counts, encoders ----
  convert_swz_kernel<<<1, 256, 0, stream>>>(P[0], w_enc_e1, 32, 32);
  convert_swz_kernel<<<1, 256, 0, stream>>>(P[4], w_enc_n1, 32, 32);
  convert_swz_kernel<<<1, 256, 0, stream>>>(P[8], w_enc_g1, 32, 32);
  convert_swz_kernel<<<1, 256, 0, stream>>>(P[12], w_core_e1, 128, 128);
  convert_swz_kernel<<<1, 256, 0, stream>>>(P[16], w_core_n1, 80, 96);
  convert_swz_kernel<<<1, 256, 0, stream>>>(P[2], w_enc_e2, 16, 32);
  convert_swz_kernel<<<1, 256, 0, stream>>>(P[6], w_enc_n2, 16, 32);
  convert_swz_kernel<<<1, 256, 0, stream>>>(P[10], w_enc_g2, 16, 32);
  convert_swz_kernel<<<1, 256, 0, stream>>>(P[14], w_core_e2, 16, 32);
  convert_swz_kernel<<<1, 256, 0, stream>>>(P[18], w_core_n2, 16, 32);
  convert_swz_kernel<<<1, 256, 0, stream>>>(P[24], w_dec_e1, 16, 32);
  convert_swz_kernel<<<1, 256, 0, stream>>>(P[28], w_dec_n1, 16, 32);

  const int ncnt = N_NODES + 2 * N_GRAPHS;
  fill0_kernel<<<(ncnt + 255) / 256, 256, 0, stream>>>(cnt, ncnt);
  count_edges_kernel<<<(E_EDGES + 255) / 256, 256, 0, stream>>>(
      receivers, edge_graph, E_EDGES, cnt_rcv, cnt_eg);
  count_nodes_kernel<<<(N_NODES + 255) / 256, 256, 0, stream>>>(node_graph,
                                                                N_NODES, cnt_ng);

  encoder_kernel<<<E_EDGES / 64, 128, 0, stream>>>(x_e, E_EDGES, w_enc_e1, P[1],
                                                   w_enc_e2, P[3], e0h, ehh);
  encoder_kernel<<<(N_NODES / 16 + 3) / 4, 128, 0, stream>>>(
      x_n, N_NODES, w_enc_n1, P[5], w_enc_n2, P[7], n0h, nhh);
  encoder_kernel<<<1, 128, 0, stream>>>(x_g, N_GRAPHS, w_enc_g1, P[9], w_enc_g2,
                                        P[11], g0h, ghh);

  // ---- message-passing steps ----
  float* out = (float*)d_out;
  float* out_e_base = out;                                    // [4][E][2]
  float* out_n_base = out + (size_t)NSTEPS * E_EDGES * 2;     // [4][N][1]
  float* out_g_base = out_n_base + (size_t)NSTEPS * N_NODES;  // [4][G][3]
  const int nacc = N_NODES * 16 + 2 * N_GRAPHS * 16;
  for (int s = 0; s < NSTEPS; ++s) {
    fill0_kernel<<<(nacc + 255) / 256, 256, 0, stream>>>(node_acc, nacc);
    edge_core_kernel<<<E_EDGES / 64, 128, 0, stream>>>(
        e0h, ehh, n0h, nhh, g0h, ghh, senders, receivers, edge_graph, w_core_e1,
        P[13], w_core_e2, P[15], w_dec_e1, P[25], P[26], P[27], node_acc, gacc_e,
        out_e_base + (size_t)s * E_EDGES * 2);
    node_core_kernel<<<(N_NODES / 16 + 3) / 4, 128, 0, stream>>>(
        n0h, nhh, g0h, ghh, node_graph, node_acc, cnt_rcv, w_core_n1, P[17],
        w_core_n2, P[19], w_dec_n1, P[29], P[30], P[31], gacc_n,
        out_n_base + (size_t)s * N_NODES);
    graph_core_kernel<<<1, 64, 0, stream>>>(
        g0h, ghh, gacc_e, cnt_eg, gacc_n, cnt_ng, P[20], P[21], P[22], P[23],
        P[32], P[33], P[34], P[35], out_g_base + (size_t)s * N_GRAPHS * 3);
  }
}
